// CodeRecursiveNeuralNetworks_20023137534858
// MI455X (gfx1250) — matrix operand order repaired and checked by reading, not compiled, not run on hardware
//
#include <hip/hip_runtime.h>
#include <hip/hip_bf16.h>

typedef __attribute__((ext_vector_type(16))) _Float16 v16h;
typedef __attribute__((ext_vector_type(8)))  _Float16 v8h;
typedef __attribute__((ext_vector_type(8)))  float    v8f;

#define TREES 256
#define MNODES 1023
#define HDIM 128
#define NCLASS 6

// Fast tanh: 1 - 2/(e^{2x}+1).  v_exp_f32 + v_rcp_f32; exact limits at +-inf.
__device__ __forceinline__ float fast_tanh(float x) {
    float e = __expf(2.0f * x);
    return 1.0f - 2.0f * __builtin_amdgcn_rcpf(e + 1.0f);
}

// ---------------------------------------------------------------------------
// Kernel 1: pre-swizzle Wh (f32 row-major [k][n]) into the exact per-lane
// v_wmma_f32_16x16x32_f16 B-fragment layout:
//   WhB[((kt*8 + ntile)*32 + lane)*16 + e] = Wh[kt*32 + (lane>=16?8:0) + (e&7) + (e>=8?16:0)][ntile*16 + (lane&15)]
// 16384 f16 = 32KB, L2-resident for every level.
// ---------------------------------------------------------------------------
__global__ void crnn_pack_whb(const float* __restrict__ Wh, _Float16* __restrict__ whb) {
    int e = blockIdx.x * 256 + threadIdx.x;           // 0..16383
    int hidx  = e & 15;
    int lane  = (e >> 4) & 31;
    int ntile = (e >> 9) & 7;
    int kt    = e >> 12;
    int k = kt * 32 + ((lane >= 16) ? 8 : 0) + (hidx & 7) + ((hidx >= 8) ? 16 : 0);
    int n = ntile * 16 + (lane & 15);
    whb[e] = (_Float16)Wh[k * HDIM + n];
}

// ---------------------------------------------------------------------------
// Kernel 2: h_init = tanh(E[node_type]) for every node (only depth-9 leaves
// matter; internal nodes are overwritten bottom-up, matching the reference).
// ---------------------------------------------------------------------------
__global__ void crnn_init_h(const int* __restrict__ node_type,
                            const float* __restrict__ E,
                            _Float16* __restrict__ h16, int total) {
    int idx = blockIdx.x * 256 + threadIdx.x;
    if (idx >= total) return;
    int node = idx >> 7;
    int col  = idx & 127;
    int t = node_type[node];
    h16[idx] = (_Float16)fast_tanh(E[t * HDIM + col]);
}

// ---------------------------------------------------------------------------
// Kernel 3: one level of the recursion. ONE WAVE computes a 16-parent x 128-col
// tile. All 4 A fragments (S = h[c0]+h[c1], siblings contiguous -> b128 loads
// + v_pk_add_f16) are loaded ONCE (32 VGPRs). Column tiles iterate outermost
// with a single live v8f accumulator each (no spills): 8 tiles x 4
// v_wmma_f32_16x16x32_f16 = 32 WMMAs per wave. Epilogue per tile:
// h[parent] = tanh(E[type[parent]] + U + bh), stored f16.
// shift = d-1; parents at depth d-1, children at depth d.
// ---------------------------------------------------------------------------
__global__ void crnn_level(_Float16* __restrict__ h16,
                           const _Float16* __restrict__ whb,
                           const int* __restrict__ node_type,
                           const float* __restrict__ E,
                           const float* __restrict__ bh,
                           int shift) {
    const int lane   = threadIdx.x;
    const int laneHi = lane >> 4;          // 0 or 1
    const int rowIn  = lane & 15;
    const int r0     = blockIdx.x * 16;    // first parent row of this tile
    const int mask   = (1 << shift) - 1;

    // A operand: row (r0 + rowIn)'s two children are contiguous nodes
    // 2^(shift+1)-1 + 2q and +1 within the tree.
    const int r    = r0 + rowIn;
    const int tree = r >> shift;
    const int q    = r & mask;
    const long long c0 = (long long)tree * MNODES + ((2 << shift) - 1) + 2 * q;
    const _Float16* h0 = h16 + c0 * HDIM;
    const _Float16* h1 = h0 + HDIM;

    // Load all 4 A fragments once (reused by all 8 column tiles).
    v16h A[4];
    #pragma unroll
    for (int kt = 0; kt < 4; ++kt) {
        const int kb = kt * 32 + laneHi * 8;
        v8h x0 = *(const v8h*)(h0 + kb);
        v8h x1 = *(const v8h*)(h0 + kb + 16);
        v8h y0 = *(const v8h*)(h1 + kb);
        v8h y1 = *(const v8h*)(h1 + kb + 16);
        v8h lo = x0 + y0;                  // v_pk_add_f16
        v8h hi = x1 + y1;
        A[kt] = __builtin_shufflevector(lo, hi, 0,1,2,3,4,5,6,7,8,9,10,11,12,13,14,15);
    }

    // Precompute epilogue row indices: lane covers rows m = laneHi*8 + v.
    int pn[8];   // parent node index (fits in i32)
    int ty[8];   // node_type[pn]
    #pragma unroll
    for (int v = 0; v < 8; ++v) {
        const int rr = r0 + laneHi * 8 + v;
        const int tr = rr >> shift;
        const int qq = rr & mask;
        pn[v] = tr * MNODES + ((1 << shift) - 1) + qq;
        ty[v] = node_type[pn[v]];
    }

    const _Float16* bbase = whb + lane * 16;
    #pragma unroll
    for (int nt = 0; nt < 8; ++nt) {
        v8f acc = {};
        #pragma unroll
        for (int kt = 0; kt < 4; ++kt) {
            v16h B = *(const v16h*)(bbase + (kt * 8 + nt) * (32 * 16));
            acc = __builtin_amdgcn_wmma_f32_16x16x32_f16(
                /*neg_a=*/false, A[kt], /*neg_b=*/false, B,
                /*c_mod=*/(short)0, acc, /*reuse_a=*/false, /*reuse_b=*/false);
        }
        // Epilogue for this tile: lane holds column nt*16 + rowIn.
        const int col = nt * 16 + rowIn;
        const float bv = bh[col];
        #pragma unroll
        for (int v = 0; v < 8; ++v) {
            const float xv = E[ty[v] * HDIM + col];
            const float val = fast_tanh(xv + acc[v] + bv);
            h16[(long long)pn[v] * HDIM + col] = (_Float16)val;
        }
    }
}

// ---------------------------------------------------------------------------
// Kernel 4: classifier head. One thread per tree: logits = h_root @ Wc + bc,
// then numerically-stable log_softmax over 6 classes.
// ---------------------------------------------------------------------------
__global__ void crnn_head(const _Float16* __restrict__ h16,
                          const int* __restrict__ root_idx,
                          const float* __restrict__ Wc,
                          const float* __restrict__ bc,
                          float* __restrict__ out) {
    int b = blockIdx.x * 256 + threadIdx.x;
    if (b >= TREES) return;
    const long long root = root_idx[b];
    const _Float16* hr = h16 + root * HDIM;
    float logit[NCLASS];
    #pragma unroll
    for (int c = 0; c < NCLASS; ++c) logit[c] = bc[c];
    for (int k = 0; k < HDIM; ++k) {
        float hv = (float)hr[k];
        #pragma unroll
        for (int c = 0; c < NCLASS; ++c) logit[c] += hv * Wc[k * NCLASS + c];
    }
    float mx = logit[0];
    #pragma unroll
    for (int c = 1; c < NCLASS; ++c) mx = fmaxf(mx, logit[c]);
    float s = 0.f;
    #pragma unroll
    for (int c = 0; c < NCLASS; ++c) s += __expf(logit[c] - mx);
    float lse = mx + __logf(s);
    #pragma unroll
    for (int c = 0; c < NCLASS; ++c) out[b * NCLASS + c] = logit[c] - lse;
}

extern "C" void kernel_launch(void* const* d_in, const int* in_sizes, int n_in,
                              void* d_out, int out_size, void* d_ws, size_t ws_size,
                              hipStream_t stream) {
    const int*   node_type = (const int*)d_in[0];
    // d_in[1] parent_idx, d_in[2] depth: topology is a fixed complete binary
    // tree, derived analytically instead.
    const int*   root_idx  = (const int*)d_in[3];
    const float* E         = (const float*)d_in[4];
    const float* Wh        = (const float*)d_in[5];
    const float* bh        = (const float*)d_in[6];
    const float* Wc        = (const float*)d_in[7];
    const float* bc        = (const float*)d_in[8];
    float* out = (float*)d_out;

    const int N = in_sizes[0];                 // 261888 nodes

    // Workspace layout: [0,32KB) swizzled WhB fragments; [64KB, ...) h in f16.
    _Float16* whb = (_Float16*)d_ws;
    _Float16* h16 = (_Float16*)((char*)d_ws + 65536);

    // 1) Pack Wh into WMMA B-fragment order (16384 elems).
    crnn_pack_whb<<<64, 256, 0, stream>>>(Wh, whb);

    // 2) h = tanh(E[node_type]) for all nodes.
    const int total = N * HDIM;
    crnn_init_h<<<(total + 255) / 256, 256, 0, stream>>>(node_type, E, h16, total);

    // 3) Bottom-up recursion: levels d = 9 .. 1 (children depth d -> parents d-1).
    for (int d = 9; d >= 1; --d) {
        const int shift = d - 1;
        const int P = TREES << shift;          // parents this level
        crnn_level<<<P / 16, 32, 0, stream>>>(h16, whb, node_type, E, bh, shift);
    }

    // 4) Classifier + log_softmax.
    crnn_head<<<1, 256, 0, stream>>>(h16, root_idx, Wc, bc, out);
}
